// SelfAttention_48120813584950
// MI455X (gfx1250) — compile-verified, matrix operand order checked by reading, not run on hardware
//
#include <hip/hip_runtime.h>

// ---------------------------------------------------------------------------
// Self-attention for MI455X (gfx1250): bf16 WMMA everywhere, f32 accumulate.
//   1) qkvo_gemm_kernel mode 0/1/2 : Q/K/V projections  (fp32 in -> bf16 out)
//        Q,K stored [n][h][s][d] ; V stored transposed [n][h][d][s]
//   2) attn_kernel               : causal flash attention, bf16 out [n][s][e]
//        K/V tiles staged with GLOBAL_LOAD_ASYNC_TO_LDS_B128 (ASYNCcnt)
//   3) qkvo_gemm_kernel mode 3   : output projection    (bf16 in -> fp32 out)
//        A-tile staged with GLOBAL_LOAD_ASYNC_TO_LDS_B128 as well
// ---------------------------------------------------------------------------

#define S_LEN  2048
#define EMB    1024
#define HEADS  16
#define HDIM   64
#define NBATCH 2
#define MROWS  (NBATCH * S_LEN)   // 4096

typedef __attribute__((ext_vector_type(16))) __bf16       v16bf;
typedef __attribute__((ext_vector_type(2)))  __bf16       v2bf;
typedef __attribute__((ext_vector_type(2)))  float        v2f;
typedef __attribute__((ext_vector_type(8)))  float        v8f;
typedef __attribute__((ext_vector_type(4)))  unsigned int v4u;

union FragU {
  v4u   q[2];
  v16bf b;
};

// f32 -> bf16 round-to-nearest-even (bit domain), scalar
__device__ __forceinline__ unsigned short f2bf(float x) {
  unsigned u = __float_as_uint(x);
  unsigned r = u + 0x7FFFu + ((u >> 16) & 1u);
  return (unsigned short)(r >> 16);
}

// Pack two f32 into packed bf16x2; prefer native packed cvt when available.
__device__ __forceinline__ unsigned pack_bf16x2(float lo, float hi) {
#if __has_builtin(__builtin_amdgcn_cvt_pk_bf16_f32)
  union { v2bf b; unsigned u; } c;
  c.b = __builtin_amdgcn_cvt_pk_bf16_f32(lo, hi);
  return c.u;
#else
  v2f f; f.x = lo; f.y = hi;
  union { v2bf b; unsigned u; } c;
  c.b = __builtin_convertvector(f, v2bf);
  return c.u;
#endif
}

// CDNA5 16-bit A/B fragment layout (16x32 A or 32x16 B, wave32):
//  lane = {idx 0..15, half 0/1}; vector elems j=0..7  hold K = 8*half + j,
//                                 elems j=8..15 hold K = 16 + 8*half + (j-8).
// => two aligned 16-byte loads from a 32-element (bf16) row.
__device__ __forceinline__ v16bf load_frag(const unsigned short* row, int half) {
  FragU f;
  f.q[0] = *(const v4u*)(row + 8 * half);
  f.q[1] = *(const v4u*)(row + 16 + 8 * half);
  return f.b;
}

__device__ __forceinline__ v8f wmma_bf16(v16bf a, v16bf b, v8f c) {
  return __builtin_amdgcn_wmma_f32_16x16x32_bf16(false, a, false, b,
                                                 (short)0, c, false, false);
}

// Async 16B global -> LDS copy (no VGPR round-trip, tracked by ASYNCcnt).
__device__ __forceinline__ void async_copy_b128(unsigned lds_off,
                                                unsigned long long gaddr) {
  asm volatile("global_load_async_to_lds_b128 %0, %1, off"
               :: "v"(lds_off), "v"(gaddr) : "memory");
}
__device__ __forceinline__ void wait_async0() {
  asm volatile("s_wait_asynccnt 0x0" ::: "memory");
}

// ---------------------------------------------------------------------------
// GEMM: Y[4096 x 1024] = X[4096 x 1024] * W[1024 x 1024] + bias
// Block tile 128(M) x 64(N), K-step 32, 256 threads = 8 waves.
// mode 0: X=f32 query  -> bf16 Q   [n][h][s][d]
// mode 1: X=f32 keys   -> bf16 K   [n][h][s][d]
// mode 2: X=f32 values -> bf16 V^T [n][h][d][s]
// mode 3: X=bf16 attn  -> f32 out  [n][s][e]   (A tile staged async)
// ---------------------------------------------------------------------------
__global__ __launch_bounds__(256) void qkvo_gemm_kernel(
    const float* __restrict__ Xf,
    const unsigned short* __restrict__ Xb,
    const float* __restrict__ W,
    const float* __restrict__ bias,
    unsigned short* __restrict__ outb,
    float* __restrict__ outf,
    int mode)
{
  __shared__ unsigned short Xs[128 * 32];   // A tile, row-major [m][k]
  __shared__ unsigned short Wt[64 * 32];    // B tile transposed [n][k]
  unsigned* Xs32 = (unsigned*)Xs;           // 2048 packed pairs
  unsigned* Wt32 = (unsigned*)Wt;           // 1024 packed pairs (k-pairs)

  const int t     = threadIdx.x;
  const int mbase = blockIdx.x * 128;
  const int nbase = blockIdx.y * 64;
  const int wv    = t >> 5;
  const int lane  = t & 31;
  const int nl    = lane & 15;
  const int half  = lane >> 4;

  const unsigned ldsX = (unsigned)(unsigned long long)(&Xs[0]);

  v8f acc[4];
#pragma unroll
  for (int g = 0; g < 4; ++g)
#pragma unroll
    for (int r = 0; r < 8; ++r) acc[g][r] = 0.0f;

  for (int kk = 0; kk < EMB; kk += 32) {
    // ---- stage A tile 128x32 ----
    if (mode == 3) {
      // bf16 source: pure byte copy -> async global->LDS (2 x b128/thread)
#pragma unroll
      for (int i = 0; i < 2; ++i) {
        int idx = i * 256 + t;            // 512 x 16B; r = row, c4 = 16B chunk
        int r = idx >> 2, c4 = idx & 3;
        unsigned long long ga = (unsigned long long)(
            Xb + (size_t)(mbase + r) * EMB + kk + c4 * 8);
        async_copy_b128(ldsX + idx * 16, ga);
      }
    } else {
      // f32 source: load float2, packed-convert, store b32
#pragma unroll
      for (int i = 0; i < 8; ++i) {
        int idx = i * 256 + t;            // idx = r*16 + c2
        int r = idx >> 4, c2 = idx & 15;
        float2 xv =
            *(const float2*)&Xf[(size_t)(mbase + r) * EMB + kk + 2 * c2];
        Xs32[idx] = pack_bf16x2(xv.x, xv.y);
      }
    }
    // ---- stage B tile 32x64 transposed -> Wt[64][32] (pairs along k) ----
#pragma unroll
    for (int i = 0; i < 4; ++i) {
      int idx = i * 256 + t;              // 1024 pairs: r2 0..15, c 0..63
      int r2 = idx >> 6, c = idx & 63;
      const float* wp = &W[(size_t)(kk + 2 * r2) * EMB + nbase + c];
      Wt32[c * 16 + r2] = pack_bf16x2(wp[0], wp[EMB]);
    }
    wait_async0();                        // no-op unless mode 3
    __syncthreads();

    v16bf a = load_frag(&Xs[(wv * 16 + nl) * 32], half);
#pragma unroll
    for (int g = 0; g < 4; ++g) {
      v16bf b = load_frag(&Wt[(g * 16 + nl) * 32], half);
      acc[g] = wmma_bf16(a, b, acc[g]);
    }
    __syncthreads();
  }

  // ---- epilogue: bias + layout scatter ----
  const int gm = mbase + wv * 16;
#pragma unroll
  for (int g = 0; g < 4; ++g) {
    const int gn = nbase + g * 16 + nl;
    const float bb = bias[gn];
#pragma unroll
    for (int r = 0; r < 8; ++r) {
      const int gmr = gm + r + 8 * half;   // C layout: M = r + 8*half
      const float val = acc[g][r] + bb;
      if (mode == 3) {
        // final output: streaming store, keep L2 for attention working set
        __builtin_nontemporal_store(val, &outf[(size_t)gmr * EMB + gn]);
      } else {
        const int n = gmr >> 11;
        const int s = gmr & (S_LEN - 1);
        const int h = gn >> 6;
        const int d = gn & (HDIM - 1);
        if (mode == 2)
          outb[(((size_t)n * HEADS + h) * HDIM + d) * S_LEN + s] = f2bf(val);
        else
          outb[(((size_t)n * HEADS + h) * S_LEN + s) * HDIM + d] = f2bf(val);
      }
    }
  }
}

// ---------------------------------------------------------------------------
// Causal flash attention.
// Block = 128 threads = 4 waves; block owns 64 query rows of one (n,h);
// wave w owns rows q0 = bx*64 + w*16 .. +15.
// K-tiles of 32 columns with causal block skip (tiles 0 .. 2*bx+1).
// Per tile: 4 WMMAs (energy), online softmax, P transpose via LDS,
//           4 WMMAs (O += P*V). K/V tiles staged via async-to-LDS.
// ---------------------------------------------------------------------------
__global__ __launch_bounds__(128) void attn_kernel(
    const unsigned short* __restrict__ qb,   // [n][h][s][d] bf16
    const unsigned short* __restrict__ kb,   // [n][h][s][d] bf16
    const unsigned short* __restrict__ vtb,  // [n][h][d][s] bf16
    unsigned short* __restrict__ attn)       // [n][s][h*64+d] bf16
{
  __shared__ unsigned short Ks[32 * 64];     // K tile [k][d]
  __shared__ unsigned short Vs[64 * 32];     // V tile [d][k]
  __shared__ unsigned short Ps[4 * 16 * 32]; // per-wave P transpose scratch

  const int t    = threadIdx.x;
  const int wv   = t >> 5;
  const int lane = t & 31;
  const int nl   = lane & 15;
  const int half = lane >> 4;
  const int bx   = blockIdx.x;
  const int h    = blockIdx.y;
  const int n    = blockIdx.z;

  const int q0 = bx * 64 + wv * 16;

  // Q fragments (persistent): A 16x32 for d=0..31 and d=32..63
  const unsigned short* qrow =
      &qb[(((size_t)n * HEADS + h) * S_LEN + q0 + nl) * HDIM];
  const v16bf aq0 = load_frag(qrow, half);
  const v16bf aq1 = load_frag(qrow + 32, half);

  float rowmax[8], rowsum[8];
  v8f o[4];
#pragma unroll
  for (int r = 0; r < 8; ++r) { rowmax[r] = -3.0e38f; rowsum[r] = 0.0f; }
#pragma unroll
  for (int g = 0; g < 4; ++g)
#pragma unroll
    for (int r = 0; r < 8; ++r) o[g][r] = 0.0f;

  const float sc = 0.125f;  // 1/sqrt(64)
  const int nkt = bx * 2 + 2;
  const unsigned short* kbbase = &kb[(((size_t)n * HEADS + h) * S_LEN) * HDIM];
  const unsigned short* vbase  = &vtb[(((size_t)n * HEADS + h) * HDIM) * S_LEN];

  const unsigned ldsK = (unsigned)(unsigned long long)(&Ks[0]);
  const unsigned ldsV = (unsigned)(unsigned long long)(&Vs[0]);

  for (int kt = 0; kt < nkt; ++kt) {
    const int kbase = kt * 32;

    // ---- async stage K tile: 4KB contiguous (2 x b128 per thread) ----
    {
      unsigned long long ga =
          (unsigned long long)(kbbase + (size_t)kbase * HDIM) +
          (unsigned)t * 16u;
      async_copy_b128(ldsK + t * 16, ga);
      async_copy_b128(ldsK + t * 16 + 2048, ga + 2048);
    }
    // ---- async stage V tile Vs[64][32] from V^T rows (stride S) ----
#pragma unroll
    for (int i = 0; i < 2; ++i) {
      int idx = i * 128 + t;
      int r = idx >> 2, c4 = idx & 3;
      unsigned long long ga =
          (unsigned long long)(vbase + (size_t)r * S_LEN + kbase + c4 * 8);
      async_copy_b128(ldsV + idx * 16, ga);
    }
    wait_async0();
    __syncthreads();

    // ---- energy tile 16x32: e0 = cols 0..15, e1 = cols 16..31 ----
    v8f e0, e1;
#pragma unroll
    for (int r = 0; r < 8; ++r) { e0[r] = 0.0f; e1[r] = 0.0f; }
    {
      const unsigned short* kr0 = &Ks[nl * 64];
      const unsigned short* kr1 = &Ks[(16 + nl) * 64];
      e0 = wmma_bf16(aq0, load_frag(kr0, half),      e0);
      e0 = wmma_bf16(aq1, load_frag(kr0 + 32, half), e0);
      e1 = wmma_bf16(aq0, load_frag(kr1, half),      e1);
      e1 = wmma_bf16(aq1, load_frag(kr1 + 32, half), e1);
    }

    // ---- online softmax (rows live across 16-lane halves) ----
    unsigned short* pw = &Ps[wv * 512];
    const int k0i = kbase + nl;
    const int k1i = kbase + 16 + nl;
#pragma unroll
    for (int r = 0; r < 8; ++r) {
      const int qi = q0 + r + 8 * half;
      float v0 = (k0i <= qi) ? e0[r] * sc : -1.0e30f;
      float v1 = (k1i <= qi) ? e1[r] * sc : -1.0e30f;
      float mt = fmaxf(v0, v1);
#pragma unroll
      for (int off = 8; off >= 1; off >>= 1)
        mt = fmaxf(mt, __shfl_xor(mt, off, 16));
      const float mnew = fmaxf(rowmax[r], mt);
      const float corr = __expf(rowmax[r] - mnew);
      const float p0 = __expf(v0 - mnew);
      const float p1 = __expf(v1 - mnew);
      float ps = p0 + p1;
#pragma unroll
      for (int off = 8; off >= 1; off >>= 1)
        ps += __shfl_xor(ps, off, 16);
      rowsum[r] = rowsum[r] * corr + ps;
      rowmax[r] = mnew;
#pragma unroll
      for (int g = 0; g < 4; ++g) o[g][r] *= corr;
      pw[(r + 8 * half) * 32 + nl]      = f2bf(p0);
      pw[(r + 8 * half) * 32 + 16 + nl] = f2bf(p1);
    }

    // ---- O += P(16x32) * V(32x64) ----
    const v16bf ap = load_frag(&pw[nl * 32], half);
#pragma unroll
    for (int g = 0; g < 4; ++g) {
      v16bf bvf = load_frag(&Vs[(g * 16 + nl) * 32], half);
      o[g] = wmma_bf16(ap, bvf, o[g]);
    }
    __syncthreads();
  }

  // ---- normalize + store bf16 [n][s][h*64+d] ----
#pragma unroll
  for (int g = 0; g < 4; ++g) {
    const int d = g * 16 + nl;
#pragma unroll
    for (int r = 0; r < 8; ++r) {
      const int s = q0 + r + 8 * half;
      const float val = o[g][r] / rowsum[r];
      attn[((size_t)n * S_LEN + s) * EMB + h * HDIM + d] = f2bf(val);
    }
  }
}

// ---------------------------------------------------------------------------
extern "C" void kernel_launch(void* const* d_in, const int* in_sizes, int n_in,
                              void* d_out, int out_size, void* d_ws, size_t ws_size,
                              hipStream_t stream) {
  const float* values = (const float*)d_in[0];
  const float* keys   = (const float*)d_in[1];
  const float* query  = (const float*)d_in[2];
  // d_in[3] = mask: causal, handled analytically in-kernel
  const float* Wv = (const float*)d_in[4];
  const float* bv = (const float*)d_in[5];
  const float* Wk = (const float*)d_in[6];
  const float* bk = (const float*)d_in[7];
  const float* Wq = (const float*)d_in[8];
  const float* bq = (const float*)d_in[9];
  const float* Wo = (const float*)d_in[10];
  const float* bo = (const float*)d_in[11];
  float* out = (float*)d_out;

  const size_t perBuf = (size_t)NBATCH * S_LEN * EMB * sizeof(unsigned short); // 8MB
  unsigned short* qb   = (unsigned short*)d_ws;
  unsigned short* kbp  = (unsigned short*)((char*)d_ws + 1 * perBuf);
  unsigned short* vtb  = (unsigned short*)((char*)d_ws + 2 * perBuf);
  unsigned short* attn = (unsigned short*)((char*)d_ws + 3 * perBuf);

  dim3 gblk(256), ggrid(MROWS / 128, EMB / 64);  // 32 x 16
  qkvo_gemm_kernel<<<ggrid, gblk, 0, stream>>>(
      query, (const unsigned short*)nullptr, Wq, bq, qb, (float*)nullptr, 0);
  qkvo_gemm_kernel<<<ggrid, gblk, 0, stream>>>(
      keys, (const unsigned short*)nullptr, Wk, bk, kbp, (float*)nullptr, 1);
  qkvo_gemm_kernel<<<ggrid, gblk, 0, stream>>>(
      values, (const unsigned short*)nullptr, Wv, bv, vtb, (float*)nullptr, 2);

  dim3 ablk(128), agrid(S_LEN / 64, HEADS, NBATCH);
  attn_kernel<<<agrid, ablk, 0, stream>>>(qb, kbp, vtb, attn);

  qkvo_gemm_kernel<<<ggrid, gblk, 0, stream>>>(
      (const float*)nullptr, attn, Wo, bo, (unsigned short*)nullptr, out, 3);
}
